// CrossAttentionBlock_34780645163445
// MI455X (gfx1250) — matrix operand order repaired and checked by reading, not compiled, not run on hardware
//
#include <hip/hip_runtime.h>
#include <hip/hip_fp16.h>

#define B_      16
#define C_      512
#define HW_     4096
#define S_      77
#define SP_     80      // S padded to 5 WMMA tiles
#define SPP_    96      // S padded for PV k-dim (3 x K32)
#define CTX_    768
#define HEADS_  8
#define HD_     64
#define GROUPS_ 32
#define EPS_    1e-5f

typedef __attribute__((ext_vector_type(16))) _Float16 v16h;
typedef __attribute__((ext_vector_type(8)))  float    v8f;

union Frag { v16h v; unsigned int u[8]; _Float16 h[16]; };

static __device__ __forceinline__ v8f wmma_f16(const Frag& a, const Frag& b, v8f c) {
  return __builtin_amdgcn_wmma_f32_16x16x32_f16(false, a.v, false, b.v, (short)0, c, false, false);
}

// ---------------------------------------------------------------- TDM support
#if defined(__has_builtin)
#if __has_builtin(__builtin_amdgcn_tensor_load_to_lds)
#define HAVE_TDM 1
#else
#define HAVE_TDM 0
#endif
#else
#define HAVE_TDM 0
#endif

#define LDSOFF(p) ((unsigned)(size_t)(p))

#if HAVE_TDM
typedef __attribute__((ext_vector_type(4))) unsigned int v4u;
typedef __attribute__((ext_vector_type(8))) int v8i;
typedef __attribute__((ext_vector_type(4))) int v4i;

// 2D tile load: tile_w elements per row, tile_h rows, row pitch stride_elems
// (f16 elements). HW pad feature reproduces the +1-dword bank-conflict pad:
//   pad_interval_code: dwords/row (3=16dw, 4=32dw, 5=64dw), pad_amount_code 0 = 1 dword.
static __device__ __forceinline__ void tdm_load_2d(const void* gptr, unsigned lds_off,
                                                   unsigned tile_w, unsigned tile_h,
                                                   unsigned stride_elems,
                                                   unsigned pad_interval_code,
                                                   unsigned pad_amount_code) {
  unsigned long long ga = (unsigned long long)(size_t)gptr;
  v4u g0 = { 1u,                                   // count=1 (valid user D#)
             lds_off,                              // lds_addr
             (unsigned)ga,                         // global_addr[31:0]
             (unsigned)(ga >> 32) | (2u << 30) };  // global_addr[56:32] | type=2
  unsigned w0 = (1u << 16)                         // data_size = 2 bytes
              | (1u << 20)                         // pad_enable
              | (pad_interval_code << 22)
              | (pad_amount_code << 25);
  unsigned td0 = stride_elems;                     // tensor_dim0 >= tile_w
  unsigned td1 = 0x40000000u;                      // tensor_dim1: effectively unbounded
  v8i g1 = { (int)w0,
             (int)((td0 & 0xffffu) << 16),                    // [63:48]  td0 lo
             (int)((td0 >> 16) | ((td1 & 0xffffu) << 16)),    // [79:64] td0 hi | [95:80] td1 lo
             (int)((td1 >> 16) | (tile_w << 16)),             // [111:96] td1 hi | [127:112] tile_dim0
             (int)(tile_h & 0xffffu),                         // [143:128] tile_dim1, tile_dim2=0
             (int)stride_elems,                               // tensor_dim0_stride[31:0]
             0, 0 };
  v4i gz = { 0, 0, 0, 0 };
#if defined(__clang_major__) && (__clang_major__ >= 23)
  v8i gz8 = { 0, 0, 0, 0, 0, 0, 0, 0 };
  __builtin_amdgcn_tensor_load_to_lds(g0, g1, gz, gz, gz8, 0);
#else
  __builtin_amdgcn_tensor_load_to_lds(g0, g1, gz, gz, 0);
#endif
}
#endif

// ---------------------------------------------------------------- converts
__global__ __launch_bounds__(256) void f32_to_f16_kernel(const float* __restrict__ s,
                                                         _Float16* __restrict__ d, int n) {
  int i = blockIdx.x * 256 + threadIdx.x;
  if (i < n) d[i] = (_Float16)s[i];
}

// ---------------------------------------------------------------- group norm
__global__ __launch_bounds__(256) void groupnorm_kernel(const float* __restrict__ x,
                                                        const float* __restrict__ w,
                                                        const float* __restrict__ b,
                                                        _Float16* __restrict__ xn) {
  const int bb = blockIdx.x / GROUPS_, g = blockIdx.x % GROUPS_;
  const int CPG = C_ / GROUPS_;                 // 16
  const int N = CPG * HW_;                      // 65536
  const float* xp = x + ((size_t)bb * C_ + (size_t)g * CPG) * HW_;
  _Float16* xo = xn + ((size_t)bb * C_ + (size_t)g * CPG) * HW_;

  float s = 0.f, s2 = 0.f;
  for (int i = threadIdx.x; i < N; i += 256) { float v = xp[i]; s += v; s2 += v * v; }
  #pragma unroll
  for (int m = 16; m >= 1; m >>= 1) { s += __shfl_xor(s, m, 32); s2 += __shfl_xor(s2, m, 32); }
  __shared__ float r0[8], r1[8];
  const int lane = threadIdx.x & 31, wv = threadIdx.x >> 5;
  if (lane == 0) { r0[wv] = s; r1[wv] = s2; }
  __syncthreads();
  float ts = 0.f, ts2 = 0.f;
  #pragma unroll
  for (int i = 0; i < 8; i++) { ts += r0[i]; ts2 += r1[i]; }
  const float mean = ts / (float)N;
  const float rstd = rsqrtf(ts2 / (float)N - mean * mean + EPS_);
  for (int i = threadIdx.x; i < N; i += 256) {
    int c = g * CPG + i / HW_;
    xo[i] = (_Float16)((xp[i] - mean) * rstd * w[c] + b[c]);
  }
}

// ---------------------------------------------------------------- layer norm
__global__ __launch_bounds__(256) void layernorm_kernel(const float* __restrict__ ctx,
                                                        const float* __restrict__ w,
                                                        const float* __restrict__ b,
                                                        _Float16* __restrict__ cn) {
  const int bb = blockIdx.x / SP_, sIdx = blockIdx.x % SP_;
  _Float16* out = cn + ((size_t)bb * SP_ + sIdx) * CTX_;
  if (sIdx >= S_) {
    for (int i = threadIdx.x; i < CTX_; i += 256) out[i] = (_Float16)0.f;
    return;
  }
  const float* xp = ctx + ((size_t)bb * S_ + sIdx) * CTX_;
  float s = 0.f, s2 = 0.f;
  for (int i = threadIdx.x; i < CTX_; i += 256) { float v = xp[i]; s += v; s2 += v * v; }
  #pragma unroll
  for (int m = 16; m >= 1; m >>= 1) { s += __shfl_xor(s, m, 32); s2 += __shfl_xor(s2, m, 32); }
  __shared__ float r0[8], r1[8];
  const int lane = threadIdx.x & 31, wv = threadIdx.x >> 5;
  if (lane == 0) { r0[wv] = s; r1[wv] = s2; }
  __syncthreads();
  float ts = 0.f, ts2 = 0.f;
  #pragma unroll
  for (int i = 0; i < 8; i++) { ts += r0[i]; ts2 += r1[i]; }
  const float mean = ts / (float)CTX_;
  const float rstd = rsqrtf(ts2 / (float)CTX_ - mean * mean + EPS_);
  for (int i = threadIdx.x; i < CTX_; i += 256)
    out[i] = (_Float16)((xp[i] - mean) * rstd * w[i] + b[i]);
}

// ---------------------------------------------------------------- K/V projection
// out[b][s][o] = sum_c cn[b][s][c] * W[o][c] + bias[o]   (M=80, N=512, K=768)
__global__ __launch_bounds__(256) void kv_proj_kernel(const _Float16* __restrict__ A,
                                                      const _Float16* __restrict__ W,
                                                      const float* __restrict__ bias,
                                                      _Float16* __restrict__ out) {
  const int bb = blockIdx.z;
  const int bm0 = blockIdx.y * 16;      // s-tile (5)
  const int bn0 = blockIdx.x * 128;     // o-tile (4)
  const int tid = threadIdx.x, lane = tid & 31, wv = tid >> 5;

  __shared__ _Float16 As[2][16][34];    // [s][c]
  __shared__ _Float16 Bs[2][128][34];   // [o][c]  (B pairs contiguous along k)

  v8f acc; for (int i = 0; i < 8; i++) acc[i] = 0.f;
  const _Float16* Ab = A + ((size_t)bb * SP_ + bm0) * CTX_;

  auto stage = [&](int buf, int k0) {
#if HAVE_TDM
    if (wv == 0) {
      tdm_load_2d(Ab + k0, LDSOFF(&As[buf][0][0]), 32, 16, CTX_, 3, 0);
      tdm_load_2d(W + (size_t)bn0 * CTX_ + k0, LDSOFF(&Bs[buf][0][0]), 32, 128, CTX_, 3, 0);
    }
#else
    { int idx = tid * 2, row = idx >> 5, k = idx & 31;
      *(unsigned int*)&As[buf][row][k] =
          *(const unsigned int*)(Ab + (size_t)row * CTX_ + k0 + k); }
    { int n = tid >> 1, half = tid & 1;
      const _Float16* gp = W + (size_t)(bn0 + n) * CTX_ + k0 + half * 16;
      _Float16 tmp[16];
      *(uint4*)&tmp[0] = *(const uint4*)gp;
      *(uint4*)&tmp[8] = *(const uint4*)(gp + 8);
      #pragma unroll
      for (int j = 0; j < 8; j++)
        *(unsigned int*)&Bs[buf][n][half * 16 + 2 * j] = *(unsigned int*)&tmp[2 * j]; }
#endif
  };

  const int NK = CTX_ / 32;             // 24
  stage(0, 0);
#if HAVE_TDM
  if (wv == 0) __builtin_amdgcn_s_wait_tensorcnt(0);
#endif
  __syncthreads();

  for (int ki = 0; ki < NK; ki++) {
    const int cur = ki & 1;
    if (ki + 1 < NK) stage(cur ^ 1, (ki + 1) * 32);

    Frag af, bf;
    { int row = lane & 15, koff = (lane >> 4) * 16;
      #pragma unroll
      for (int j = 0; j < 8; j++) {
        int k = 2 * j + koff;
        af.u[j] = *(const unsigned int*)&As[cur][row][k]; } }
    { int col = wv * 16 + (lane & 15), kb = (lane >> 4) * 16;
      #pragma unroll
      for (int j = 0; j < 8; j++)
        bf.u[j] = *(const unsigned int*)&Bs[cur][col][kb + 2 * j]; }
    acc = wmma_f16(af, bf, acc);

#if HAVE_TDM
    if (wv == 0 && ki + 1 < NK) __builtin_amdgcn_s_wait_tensorcnt(0);
#endif
    __syncthreads();
  }

  const int m0 = 8 * (lane >> 4);
  const int ocol = bn0 + wv * 16 + (lane & 15);
  const float bo = bias[ocol];
  #pragma unroll
  for (int r = 0; r < 8; r++) {
    int sRow = bm0 + m0 + r;
    out[((size_t)bb * SP_ + sRow) * C_ + ocol] = (_Float16)(acc[r] + bo);
  }
}

// ---------------------------------------------------------------- main projection GEMM
// out[b][o][n] = sum_c W[o][c] * X[b][c][n] + bias[o] (+residual, f32 out)
template <int OUT_F32>
__global__ __launch_bounds__(256) void proj_gemm_kernel(const _Float16* __restrict__ W,
                                                        const _Float16* __restrict__ X,
                                                        const float* __restrict__ bias,
                                                        const float* __restrict__ residual,
                                                        _Float16* __restrict__ outh,
                                                        float* __restrict__ outf) {
  const int bb = blockIdx.z;
  const int bm0 = blockIdx.y * 64;      // o
  const int bn0 = blockIdx.x * 128;     // hw
  const int tid = threadIdx.x, lane = tid & 31, wv = tid >> 5;
  const int wm = wv & 3, wn = wv >> 2;

  __shared__ _Float16 As[2][64][34];    // [o][c]  row-major, +1 dword pad
  __shared__ _Float16 Bs[2][32][130];   // [c][n]  row-major, +1 dword pad

  v8f acc[4];
  #pragma unroll
  for (int t = 0; t < 4; t++) for (int i = 0; i < 8; i++) acc[t][i] = 0.f;

  const _Float16* Xb = X + (size_t)bb * C_ * HW_;

  auto stage = [&](int buf, int k0) {
#if HAVE_TDM
    if (wv == 0) {
      tdm_load_2d(W + (size_t)bm0 * C_ + k0, LDSOFF(&As[buf][0][0]), 32, 64, C_, 3, 0);
      tdm_load_2d(Xb + (size_t)k0 * HW_ + bn0, LDSOFF(&Bs[buf][0][0]), 128, 32, HW_, 5, 0);
    }
#else
    { int row = tid >> 2, seg = tid & 3;
      const _Float16* gp = W + (size_t)(bm0 + row) * C_ + k0 + seg * 8;
      _Float16 tmp[8]; *(uint4*)tmp = *(const uint4*)gp;
      #pragma unroll
      for (int j = 0; j < 4; j++)
        *(unsigned int*)&As[buf][row][seg * 8 + 2 * j] = *(unsigned int*)&tmp[2 * j]; }
    { int kr = tid >> 3, seg = tid & 7;
      const _Float16* gp = Xb + (size_t)(k0 + kr) * HW_ + bn0 + seg * 16;
      _Float16 tmp[16];
      *(uint4*)&tmp[0] = *(const uint4*)gp;
      *(uint4*)&tmp[8] = *(const uint4*)(gp + 8);
      #pragma unroll
      for (int j = 0; j < 8; j++)
        *(unsigned int*)&Bs[buf][kr][seg * 16 + 2 * j] = *(unsigned int*)&tmp[2 * j]; }
#endif
  };

  const int NK = C_ / 32;               // 16
  stage(0, 0);
#if HAVE_TDM
  if (wv == 0) __builtin_amdgcn_s_wait_tensorcnt(0);
#endif
  __syncthreads();

  for (int ki = 0; ki < NK; ki++) {
    const int cur = ki & 1;
    if (ki + 1 < NK) stage(cur ^ 1, (ki + 1) * 32);

    Frag af;
    { int row = wm * 16 + (lane & 15), koff = (lane >> 4) * 16;
      #pragma unroll
      for (int j = 0; j < 8; j++) {
        int k = 2 * j + koff;
        af.u[j] = *(const unsigned int*)&As[cur][row][k]; } }
    #pragma unroll
    for (int t = 0; t < 4; t++) {
      Frag bf;
      int col = wn * 64 + t * 16 + (lane & 15), kb = (lane >> 4) * 16;
      #pragma unroll
      for (int j = 0; j < 8; j++) {                // paired u16 reads: rows k, k+1
        bf.h[2 * j]     = Bs[cur][kb + 2 * j][col];
        bf.h[2 * j + 1] = Bs[cur][kb + 2 * j + 1][col];
      }
      acc[t] = wmma_f16(af, bf, acc[t]);
    }

#if HAVE_TDM
    if (wv == 0 && ki + 1 < NK) __builtin_amdgcn_s_wait_tensorcnt(0);
#endif
    __syncthreads();
  }

  const int m0 = 8 * (lane >> 4);
  const int orow0 = bm0 + wm * 16 + m0;
  #pragma unroll
  for (int t = 0; t < 4; t++) {
    int ng = bn0 + wn * 64 + t * 16 + (lane & 15);
    #pragma unroll
    for (int r = 0; r < 8; r++) {
      int orow = orow0 + r;
      size_t idx = ((size_t)bb * C_ + orow) * HW_ + ng;
      float v = acc[t][r] + bias[orow];
      if (OUT_F32) outf[idx] = v + residual[idx];
      else         outh[idx] = (_Float16)v;
    }
  }
}

// ---------------------------------------------------------------- fused attention
// per (b,h): scores = Q(4096x64) K^T * 1/8 -> softmax(77) -> @ V(77x64)
__global__ __launch_bounds__(256) void attention_kernel(const _Float16* __restrict__ qg,  // [B][C][HW]
                                                        const _Float16* __restrict__ kx,  // [B][SP][C]
                                                        const _Float16* __restrict__ vx,  // [B][SP][C]
                                                        _Float16* __restrict__ og) {      // [B][C][HW]
  const int bb = blockIdx.y / HEADS_, h = blockIdx.y % HEADS_;
  const int hw0 = blockIdx.x * 128;
  const int tid = threadIdx.x, lane = tid & 31, wv = tid >> 5;

  __shared__ _Float16 Ks[SP_][66];            // [s][d]   (QK^T B pairs along d)
  __shared__ _Float16 VsT[HD_][SPP_ + 2];     // [d][s]   (PV  B pairs along s)
  __shared__ _Float16 Ps[8][16][SPP_ + 2];    // per-wave P [q][s]

  // stage K: TDM tile 64 x 80, row pitch C_, pad -> stride 66
#if HAVE_TDM
  if (wv == 0)
    tdm_load_2d(kx + (size_t)bb * SP_ * C_ + h * HD_, LDSOFF(&Ks[0][0]), 64, SP_, C_, 4, 0);
#else
  for (int i = tid; i < SP_ * 8; i += 256) {
    int s = i >> 3, seg = i & 7;
    const _Float16* gp = kx + ((size_t)bb * SP_ + s) * C_ + h * HD_ + seg * 8;
    _Float16 tmp[8]; *(uint4*)tmp = *(const uint4*)gp;
    #pragma unroll
    for (int j = 0; j < 4; j++)
      *(unsigned int*)&Ks[s][seg * 8 + 2 * j] = *(unsigned int*)&tmp[2 * j];
  }
#endif
  // stage V^T (transpose: no TDM) + zero pad rows s in [80,96)
  for (int i = tid; i < HD_ * 16; i += 256) VsT[i >> 4][SP_ + (i & 15)] = (_Float16)0.f;
  for (int i = tid; i < SP_ * 8; i += 256) {
    int s = i >> 3, seg = i & 7;
    const _Float16* gp = vx + ((size_t)bb * SP_ + s) * C_ + h * HD_ + seg * 8;
    _Float16 tmp[8]; *(uint4*)tmp = *(const uint4*)gp;
    #pragma unroll
    for (int j = 0; j < 8; j++) VsT[seg * 8 + j][s] = tmp[j];
  }
#if HAVE_TDM
  if (wv == 0) __builtin_amdgcn_s_wait_tensorcnt(0);
#endif
  __syncthreads();

  // Q A-fragments straight from global (row = query, k = d; pairs stride HW apart)
  const int q0 = wv * 16;
  Frag aq[2];
  { const int row = hw0 + q0 + (lane & 15), koff = (lane >> 4) * 16;
    const _Float16* qb = qg + ((size_t)bb * C_ + h * HD_) * HW_ + row;
    #pragma unroll
    for (int ks = 0; ks < 2; ks++)
      #pragma unroll
      for (int j = 0; j < 8; j++) {
        int d0 = ks * 32 + 2 * j + koff;
        aq[ks].h[2 * j]     = qb[(size_t)d0 * HW_];
        aq[ks].h[2 * j + 1] = qb[(size_t)(d0 + 1) * HW_];
      } }

  // scores: 5 s-tiles x 2 k-steps
  v8f accS[5];
  #pragma unroll
  for (int t = 0; t < 5; t++) for (int i = 0; i < 8; i++) accS[t][i] = 0.f;
  #pragma unroll
  for (int t = 0; t < 5; t++) {
    #pragma unroll
    for (int ks = 0; ks < 2; ks++) {
      Frag bk;
      int col = t * 16 + (lane & 15);
      int kb = (lane >> 4) * 16 + ks * 32;
      #pragma unroll
      for (int j = 0; j < 8; j++)
        bk.u[j] = *(const unsigned int*)&Ks[col][kb + 2 * j];
      accS[t] = wmma_f16(aq[ks], bk, accS[t]);
    }
  }

  // masked softmax over 80 cols (cols in lane%16 across 5 tiles; rows in VGPR r + 8*(lane/16))
  const float scale = 0.125f;                 // HD^-0.5
  float p[5][8];
  #pragma unroll
  for (int t = 0; t < 5; t++) {
    bool valid = (t * 16 + (lane & 15)) < S_;
    #pragma unroll
    for (int r = 0; r < 8; r++) p[t][r] = valid ? accS[t][r] * scale : -1e30f;
  }
  #pragma unroll
  for (int r = 0; r < 8; r++) {
    float mx = p[0][r];
    #pragma unroll
    for (int t = 1; t < 5; t++) mx = fmaxf(mx, p[t][r]);
    #pragma unroll
    for (int off = 8; off >= 1; off >>= 1) mx = fmaxf(mx, __shfl_xor(mx, off, 32));
    float sm = 0.f;
    #pragma unroll
    for (int t = 0; t < 5; t++) { float e = __expf(p[t][r] - mx); p[t][r] = e; sm += e; }
    #pragma unroll
    for (int off = 8; off >= 1; off >>= 1) sm += __shfl_xor(sm, off, 32);
    float inv = 1.f / sm;
    #pragma unroll
    for (int t = 0; t < 5; t++) p[t][r] *= inv;
  }

  // spill P to LDS in A layout, zero pad s in [80,96)
  { const int rb = 8 * (lane >> 4), cb = lane & 15;
    #pragma unroll
    for (int t = 0; t < 5; t++)
      #pragma unroll
      for (int r = 0; r < 8; r++) Ps[wv][rb + r][t * 16 + cb] = (_Float16)p[t][r];
    for (int i = lane; i < 256; i += 32) Ps[wv][i >> 4][SP_ + (i & 15)] = (_Float16)0.f; }
  __syncthreads();

  // out = P (16x96) @ V (96x64): 3 k-steps x 4 d-tiles
  v8f accO[4];
  #pragma unroll
  for (int t = 0; t < 4; t++) for (int i = 0; i < 8; i++) accO[t][i] = 0.f;
  #pragma unroll
  for (int ks = 0; ks < 3; ks++) {
    Frag ap;
    { int row = lane & 15, koff = (lane >> 4) * 16;
      #pragma unroll
      for (int j = 0; j < 8; j++) {
        int k = ks * 32 + 2 * j + koff;
        ap.u[j] = *(const unsigned int*)&Ps[wv][row][k]; } }
    #pragma unroll
    for (int t = 0; t < 4; t++) {
      Frag bv;
      int col = t * 16 + (lane & 15);
      int kb = (lane >> 4) * 16 + ks * 32;
      #pragma unroll
      for (int j = 0; j < 8; j++)
        bv.u[j] = *(const unsigned int*)&VsT[col][kb + 2 * j];
      accO[t] = wmma_f16(ap, bv, accO[t]);
    }
  }

  // store in [b][c=h*64+d][hw] layout for the out-projection
  const int m0 = 8 * (lane >> 4), nb = lane & 15;
  #pragma unroll
  for (int t = 0; t < 4; t++) {
    int d = t * 16 + nb;
    _Float16* op = og + ((size_t)bb * C_ + h * HD_ + d) * HW_ + hw0 + q0 + m0;
    #pragma unroll
    for (int r = 0; r < 8; r++) op[r] = (_Float16)accO[t][r];
  }
}

// ---------------------------------------------------------------- launch
extern "C" void kernel_launch(void* const* d_in, const int* in_sizes, int n_in,
                              void* d_out, int out_size, void* d_ws, size_t ws_size,
                              hipStream_t stream) {
  const float* x     = (const float*)d_in[0];
  const float* ctx   = (const float*)d_in[1];
  const float* gn_w  = (const float*)d_in[2];
  const float* gn_b  = (const float*)d_in[3];
  const float* ln_w  = (const float*)d_in[4];
  const float* ln_b  = (const float*)d_in[5];
  const float* q_w   = (const float*)d_in[6];
  const float* q_b   = (const float*)d_in[7];
  const float* k_w   = (const float*)d_in[8];
  const float* k_b   = (const float*)d_in[9];
  const float* v_w   = (const float*)d_in[10];
  const float* v_b   = (const float*)d_in[11];
  const float* out_w = (const float*)d_in[12];
  const float* out_b = (const float*)d_in[13];

  char* ws = (char*)d_ws;
  size_t off = 0;
  auto take = [&](size_t bytes) { void* p = ws + off; off += (bytes + 255) & ~(size_t)255; return p; };
  _Float16* wq  = (_Float16*)take((size_t)C_ * C_ * 2);
  _Float16* wk  = (_Float16*)take((size_t)C_ * CTX_ * 2);
  _Float16* wv  = (_Float16*)take((size_t)C_ * CTX_ * 2);
  _Float16* wo  = (_Float16*)take((size_t)C_ * C_ * 2);
  _Float16* xn  = (_Float16*)take((size_t)B_ * C_ * HW_ * 2);
  _Float16* cn  = (_Float16*)take((size_t)B_ * SP_ * CTX_ * 2);
  _Float16* kf  = (_Float16*)take((size_t)B_ * SP_ * C_ * 2);
  _Float16* vf  = (_Float16*)take((size_t)B_ * SP_ * C_ * 2);
  _Float16* qf  = (_Float16*)take((size_t)B_ * C_ * HW_ * 2);
  _Float16* att = (_Float16*)take((size_t)B_ * C_ * HW_ * 2);

  const int nWqq = C_ * C_, nWkv = C_ * CTX_;
  f32_to_f16_kernel<<<(nWqq + 255) / 256, 256, 0, stream>>>(q_w, wq, nWqq);
  f32_to_f16_kernel<<<(nWkv + 255) / 256, 256, 0, stream>>>(k_w, wk, nWkv);
  f32_to_f16_kernel<<<(nWkv + 255) / 256, 256, 0, stream>>>(v_w, wv, nWkv);
  f32_to_f16_kernel<<<(nWqq + 255) / 256, 256, 0, stream>>>(out_w, wo, nWqq);

  groupnorm_kernel<<<B_ * GROUPS_, 256, 0, stream>>>(x, gn_w, gn_b, xn);
  layernorm_kernel<<<B_ * SP_, 256, 0, stream>>>(ctx, ln_w, ln_b, cn);

  kv_proj_kernel<<<dim3(C_ / 128, SP_ / 16, B_), 256, 0, stream>>>(cn, wk, k_b, kf);
  kv_proj_kernel<<<dim3(C_ / 128, SP_ / 16, B_), 256, 0, stream>>>(cn, wv, v_b, vf);

  proj_gemm_kernel<0><<<dim3(HW_ / 128, C_ / 64, B_), 256, 0, stream>>>(
      wq, xn, q_b, nullptr, qf, nullptr);

  attention_kernel<<<dim3(HW_ / 128, B_ * HEADS_), 256, 0, stream>>>(qf, kf, vf, att);

  proj_gemm_kernel<1><<<dim3(HW_ / 128, C_ / 64, B_), 256, 0, stream>>>(
      wo, att, out_b, x, nullptr, (float*)d_out);
}
